// MySelfAttention_45612552684057
// MI455X (gfx1250) — matrix operand order repaired and checked by reading, not compiled, not run on hardware
//
#include <hip/hip_runtime.h>
#include <hip/hip_bf16.h>

typedef __bf16 bf16;
typedef __bf16 v16bf __attribute__((ext_vector_type(16)));
typedef __bf16 v8bf  __attribute__((ext_vector_type(8)));
typedef float  v8f   __attribute__((ext_vector_type(8)));
typedef int    v4i   __attribute__((ext_vector_type(4)));

#define B_      8
#define N_TOK   4096
#define C_      256
#define HEADS_  8
#define DH_     32
#define NKV     256
#define M_Q     (B_ * N_TOK)   /* 32768 */
#define M_SR    (B_ * NKV)     /* 2048  */
#define K_SR    4096

// ---- gfx1250 async-to-LDS path (guarded; manual fallback if absent) --------
#if defined(__has_builtin)
#if __has_builtin(__builtin_amdgcn_global_load_async_to_lds_b128) && \
    __has_builtin(__builtin_amdgcn_s_wait_asynccnt)
#define USE_ASYNC_LDS 1
#endif
#endif
#ifndef USE_ASYNC_LDS
#define USE_ASYNC_LDS 0
#endif

// builtin signature (from compiler diagnostic): (v4i AS1*, v4i AS3*, imm, imm)
typedef __attribute__((address_space(1))) v4i* gv4i_p;
typedef __attribute__((address_space(3))) v4i* lv4i_p;
static __device__ __forceinline__ gv4i_p as_global(const void* p) {
  return (gv4i_p)(void*)p;
}
static __device__ __forceinline__ lv4i_p as_lds(void* p) { return (lv4i_p)p; }

static __device__ __forceinline__ v16bf cat8(v8bf lo, v8bf hi) {
  v16bf r;
#pragma unroll
  for (int i = 0; i < 8; ++i) { r[i] = lo[i]; r[i + 8] = hi[i]; }
  return r;
}
static __device__ __forceinline__ v8bf lo8(v16bf x) {
  v8bf r;
#pragma unroll
  for (int i = 0; i < 8; ++i) r[i] = x[i];
  return r;
}
static __device__ __forceinline__ v8bf hi8(v16bf x) {
  v8bf r;
#pragma unroll
  for (int i = 0; i < 8; ++i) r[i] = x[i + 8];
  return r;
}

static __device__ __forceinline__ v8f wmma_bf16(v16bf a, v16bf b, v8f c) {
  return __builtin_amdgcn_wmma_f32_16x16x32_bf16(false, a, false, b, (short)0, c,
                                                 false, false);
}

// ---------------------------------------------------------------------------
// Packing / conversion kernels
// ---------------------------------------------------------------------------

// plain f32 -> bf16 (also packs torch [out][in] weights directly as B=[N][K])
__global__ void cvt_bf16_kernel(const float* __restrict__ in, bf16* __restrict__ out,
                                int n) {
  int idx = blockIdx.x * 256 + threadIdx.x;
  if (idx < n) out[idx] = (bf16)in[idx];
}

// sr_w OIHW [256][256][4][4] -> wt[o][(dy*4+dx)*256 + c]   (B = [N][K] layout)
__global__ void pack_sr_kernel(const float* __restrict__ w, bf16* __restrict__ wt) {
  int idx = blockIdx.x * 256 + threadIdx.x;  // o*4096 + p*256 + c
  int c  = idx & 255;
  int p  = (idx >> 8) & 15;
  int o  = idx >> 12;
  int dy = p >> 2, dx = p & 3;
  wt[idx] = (bf16)w[((o * 256 + c) * 4 + dy) * 4 + dx];
}

// im2col for 4x4/stride4 patch conv: out[m=(b,ph,pw)][kk=(dy,dx,c)]
__global__ void im2col_kernel(const float* __restrict__ x, bf16* __restrict__ out) {
  size_t idx = (size_t)blockIdx.x * 256 + threadIdx.x;  // 2048*4096
  int kk = (int)(idx & 4095);
  int m  = (int)(idx >> 12);
  int c  = kk & 255;
  int p  = kk >> 8;
  int dy = p >> 2, dx = p & 3;
  int b  = m >> 8;
  int pp = m & 255;
  int ph = pp >> 4, pw = pp & 15;
  int n  = (ph * 4 + dy) * 64 + (pw * 4 + dx);
  out[idx] = (bf16)x[((size_t)(b * N_TOK + n)) * C_ + c];
}

// ---------------------------------------------------------------------------
// bf16 WMMA GEMM: out[M][N] = A[M][K] * Bp[N][K]^T + bias
// Block tile 128x64, 8 waves each 32x32 (2x2 accums), K-step 32,
// double-buffered LDS; async-DMA staging when available.
// MODE 0: f32 out, MODE 1: bf16 out.
// ---------------------------------------------------------------------------
#define GTM 128
#define GTN 64
#define GTK 32
#define LDA 40
#define LDB 40

template <int MODE>
__global__ __launch_bounds__(256) void gemm_bf16_kernel(
    const bf16* __restrict__ A, const bf16* __restrict__ Bp,
    const float* __restrict__ bias, void* __restrict__ out,
    int M, int N, int K) {
  __shared__ __align__(16) bf16 Alds[2][GTM][LDA];
  __shared__ __align__(16) bf16 Blds[2][GTN][LDB];  // [n][k]

  int m0   = blockIdx.x * GTM;
  int n0   = blockIdx.y * GTN;
  int tid  = threadIdx.x;
  int wave = tid >> 5;
  int lane = tid & 31;
  int wm   = (wave & 3) * 32;
  int wn   = (wave >> 2) * 32;
  int rlo  = lane & 15;
  int sel  = lane >> 4;

  // staging coords: A tile 128x32 = 256 x 16-half chunks; B tile 64x32 = 128 chunks
  int arow = tid >> 1, akh = (tid & 1) * 16;
  int brow = (tid & 127) >> 1, bkh = (tid & 1) * 16;
  const bf16* aptr = A + (size_t)(m0 + arow) * K + akh;
  const bf16* bptr = Bp + (size_t)(n0 + brow) * K + bkh;

  v8f acc[2][2];
#pragma unroll
  for (int i = 0; i < 2; ++i)
#pragma unroll
    for (int j = 0; j < 2; ++j) acc[i][j] = (v8f){0, 0, 0, 0, 0, 0, 0, 0};

  v16bf areg, breg;

  auto fetch = [&](int k0) {
    areg = *(const v16bf*)(aptr + k0);
    if (tid < 128) breg = *(const v16bf*)(bptr + k0);
  };
  auto commit = [&](int buf) {
    *(v8bf*)&Alds[buf][arow][akh]     = lo8(areg);
    *(v8bf*)&Alds[buf][arow][akh + 8] = hi8(areg);
    if (tid < 128) {
      *(v8bf*)&Blds[buf][brow][bkh]     = lo8(breg);
      *(v8bf*)&Blds[buf][brow][bkh + 8] = hi8(breg);
    }
  };
#if USE_ASYNC_LDS
  auto stage_async = [&](int buf, int k0) {
    __builtin_amdgcn_global_load_async_to_lds_b128(
        as_global(aptr + k0), as_lds(&Alds[buf][arow][akh]), 0, 0);
    __builtin_amdgcn_global_load_async_to_lds_b128(
        as_global(aptr + k0 + 8), as_lds(&Alds[buf][arow][akh + 8]), 0, 0);
    if (tid < 128) {
      __builtin_amdgcn_global_load_async_to_lds_b128(
          as_global(bptr + k0), as_lds(&Blds[buf][brow][bkh]), 0, 0);
      __builtin_amdgcn_global_load_async_to_lds_b128(
          as_global(bptr + k0 + 8), as_lds(&Blds[buf][brow][bkh + 8]), 0, 0);
    }
  };
#endif

  auto compute = [&](int buf) {
    v16bf afrag[2], bfrag[2];
#pragma unroll
    for (int tm = 0; tm < 2; ++tm) {
      v8bf lo = *(const v8bf*)&Alds[buf][wm + tm * 16 + rlo][sel * 8];
      v8bf hi = *(const v8bf*)&Alds[buf][wm + tm * 16 + rlo][16 + sel * 8];
      afrag[tm] = cat8(lo, hi);
    }
#pragma unroll
    for (int tn = 0; tn < 2; ++tn) {
      v8bf lo = *(const v8bf*)&Blds[buf][wn + tn * 16 + rlo][sel * 8];
      v8bf hi = *(const v8bf*)&Blds[buf][wn + tn * 16 + rlo][16 + sel * 8];
      bfrag[tn] = cat8(lo, hi);
    }
#pragma unroll
    for (int tm = 0; tm < 2; ++tm)
#pragma unroll
      for (int tn = 0; tn < 2; ++tn)
        acc[tm][tn] = wmma_bf16(afrag[tm], bfrag[tn], acc[tm][tn]);
  };

  int nk = K / GTK;
#if USE_ASYNC_LDS
  stage_async(0, 0);
  __builtin_amdgcn_s_wait_asynccnt(0);
  __syncthreads();
  for (int i = 0; i < nk; ++i) {
    if (i + 1 < nk) stage_async((i + 1) & 1, (i + 1) * GTK);
    compute(i & 1);
    if (i + 1 < nk) __builtin_amdgcn_s_wait_asynccnt(0);
    __syncthreads();
  }
#else
  fetch(0);
  commit(0);
  __syncthreads();
  for (int i = 0; i < nk; ++i) {
    if (i + 1 < nk) fetch((i + 1) * GTK);
    compute(i & 1);
    if (i + 1 < nk) commit((i + 1) & 1);
    __syncthreads();
  }
#endif

#pragma unroll
  for (int tm = 0; tm < 2; ++tm)
#pragma unroll
    for (int tn = 0; tn < 2; ++tn) {
      int mbase = m0 + wm + tm * 16 + sel * 8;
      int n     = n0 + wn + tn * 16 + rlo;
      float bv  = bias[n];
#pragma unroll
      for (int r = 0; r < 8; ++r) {
        float v    = acc[tm][tn][r] + bv;
        size_t idx = (size_t)(mbase + r) * N + n;
        if (MODE == 0) ((float*)out)[idx] = v;
        else           ((bf16*)out)[idx]  = (bf16)v;
      }
    }
}

// ---------------------------------------------------------------------------
// LayerNorm over C=256, one block (256 thr) per row, bf16 out
// ---------------------------------------------------------------------------
__global__ __launch_bounds__(256) void ln_kernel(const float* __restrict__ xin,
                                                 const float* __restrict__ g,
                                                 const float* __restrict__ b,
                                                 bf16* __restrict__ xout) {
  __shared__ float red[8];
  int row = blockIdx.x, t = threadIdx.x;
  float v = xin[(size_t)row * C_ + t];
  float s = v;
#pragma unroll
  for (int o = 16; o > 0; o >>= 1) s += __shfl_xor(s, o, 32);
  if ((t & 31) == 0) red[t >> 5] = s;
  __syncthreads();
  float tot = 0.0f;
#pragma unroll
  for (int i = 0; i < 8; ++i) tot += red[i];
  float mu = tot * (1.0f / 256.0f);
  float d  = v - mu;
  float q  = d * d;
  __syncthreads();
#pragma unroll
  for (int o = 16; o > 0; o >>= 1) q += __shfl_xor(q, o, 32);
  if ((t & 31) == 0) red[t >> 5] = q;
  __syncthreads();
  float var = 0.0f;
#pragma unroll
  for (int i = 0; i < 8; ++i) var += red[i];
  var *= (1.0f / 256.0f);
  float y = d * rsqrtf(var + 1e-5f) * g[t] + b[t];
  xout[(size_t)row * C_ + t] = (bf16)y;
}

// ---------------------------------------------------------------------------
// Attention: per (b,h, 128-row block). softmax(QK^T * scale) @ V
// ---------------------------------------------------------------------------
#define LDK 40
#define VLD 264

__global__ __launch_bounds__(256) void attn_kernel(const bf16* __restrict__ qmat,
                                                   const bf16* __restrict__ kvmat,
                                                   bf16* __restrict__ omat) {
  __shared__ __align__(16) bf16 klds[NKV][LDK];      // [j][d]
  __shared__ __align__(16) bf16 vtlds[DH_][VLD];     // [d][j]
  __shared__ __align__(16) bf16 plds[8][16][LDK];    // per-wave P restage

  int bh   = blockIdx.y;  // b*8 + h
  int b    = bh >> 3, h = bh & 7;
  int tid  = threadIdx.x;
  int wave = tid >> 5;
  int lane = tid & 31;
  int rlo  = lane & 15;
  int sel  = lane >> 4;

  // stage K [256][32] (async if available) and V^T [32][256]
  {
    int j = tid;
    const bf16* krow = kvmat + (size_t)(b * NKV + j) * (2 * C_) + h * DH_;
#if USE_ASYNC_LDS
#pragma unroll
    for (int q = 0; q < 4; ++q)
      __builtin_amdgcn_global_load_async_to_lds_b128(
          as_global(krow + q * 8), as_lds(&klds[j][q * 8]), 0, 0);
#else
#pragma unroll
    for (int q = 0; q < 4; ++q)
      *(v8bf*)&klds[j][q * 8] = *(const v8bf*)(krow + q * 8);
#endif
    const bf16* vrow = krow + C_;
#pragma unroll
    for (int q = 0; q < 4; ++q) {
      v8bf vv = *(const v8bf*)(vrow + q * 8);
#pragma unroll
      for (int i = 0; i < 8; ++i) vtlds[q * 8 + i][j] = vv[i];
    }
  }
#if USE_ASYNC_LDS
  __builtin_amdgcn_s_wait_asynccnt(0);
#endif
  __syncthreads();

  // Q fragment (16 rows x K=32), A-matrix layout
  int qrow0 = b * N_TOK + blockIdx.x * 128 + wave * 16;
  const bf16* qbase = qmat + (size_t)(qrow0 + rlo) * C_ + h * DH_;
  v16bf qa = cat8(*(const v8bf*)(qbase + sel * 8),
                  *(const v8bf*)(qbase + 16 + sel * 8));

  // S = Q K^T : 16 column tiles of 16
  v8f s[16];
#pragma unroll
  for (int t = 0; t < 16; ++t) {
    v8bf lo = *(const v8bf*)&klds[t * 16 + rlo][sel * 8];
    v8bf hi = *(const v8bf*)&klds[t * 16 + rlo][16 + sel * 8];
    v8f z = (v8f){0, 0, 0, 0, 0, 0, 0, 0};
    s[t] = wmma_bf16(qa, cat8(lo, hi), z);
  }

  // softmax rows (row = r + sel*8; lane half covers 16 cols per tile)
  const float scale = 0.17677669529663687f;  // 1/sqrt(32)
#pragma unroll
  for (int r = 0; r < 8; ++r) {
    float mm = s[0][r] * scale;
#pragma unroll
    for (int t = 1; t < 16; ++t) mm = fmaxf(mm, s[t][r] * scale);
#pragma unroll
    for (int o = 8; o > 0; o >>= 1) mm = fmaxf(mm, __shfl_xor(mm, o, 16));
    float sum = 0.0f;
#pragma unroll
    for (int t = 0; t < 16; ++t) {
      float p = __expf(s[t][r] * scale - mm);
      s[t][r] = p;
      sum += p;
    }
#pragma unroll
    for (int o = 8; o > 0; o >>= 1) sum += __shfl_xor(sum, o, 16);
    float inv = 1.0f / sum;
#pragma unroll
    for (int t = 0; t < 16; ++t) s[t][r] *= inv;
  }

  // O = P @ V : restage P (C-layout -> A-layout) through per-wave LDS patch
  v8f o0 = (v8f){0, 0, 0, 0, 0, 0, 0, 0};
  v8f o1 = (v8f){0, 0, 0, 0, 0, 0, 0, 0};
#pragma unroll
  for (int t2 = 0; t2 < 8; ++t2) {
#pragma unroll
    for (int half = 0; half < 2; ++half) {
      int t = t2 * 2 + half;
#pragma unroll
      for (int r = 0; r < 8; ++r)
        plds[wave][r + sel * 8][half * 16 + rlo] = (bf16)s[t][r];
    }
    v16bf pa = cat8(*(const v8bf*)&plds[wave][rlo][sel * 8],
                    *(const v8bf*)&plds[wave][rlo][16 + sel * 8]);
    {
      v8bf lo = *(const v8bf*)&vtlds[rlo][t2 * 32 + sel * 8];
      v8bf hi = *(const v8bf*)&vtlds[rlo][t2 * 32 + 16 + sel * 8];
      o0 = wmma_bf16(pa, cat8(lo, hi), o0);
    }
    {
      v8bf lo = *(const v8bf*)&vtlds[16 + rlo][t2 * 32 + sel * 8];
      v8bf hi = *(const v8bf*)&vtlds[16 + rlo][t2 * 32 + 16 + sel * 8];
      o1 = wmma_bf16(pa, cat8(lo, hi), o1);
    }
  }

#pragma unroll
  for (int r = 0; r < 8; ++r) {
    size_t grow = (size_t)(qrow0 + r + sel * 8) * C_;
    omat[grow + h * DH_ + rlo]      = (bf16)o0[r];
    omat[grow + h * DH_ + 16 + rlo] = (bf16)o1[r];
  }
}

// ---------------------------------------------------------------------------
// Host launcher
// ---------------------------------------------------------------------------
extern "C" void kernel_launch(void* const* d_in, const int* in_sizes, int n_in,
                              void* d_out, int out_size, void* d_ws, size_t ws_size,
                              hipStream_t stream) {
  const float* x      = (const float*)d_in[0];
  const float* q_w    = (const float*)d_in[3];
  const float* q_b    = (const float*)d_in[4];
  const float* kv_w   = (const float*)d_in[5];
  const float* kv_b   = (const float*)d_in[6];
  const float* sr_w   = (const float*)d_in[7];
  const float* sr_b   = (const float*)d_in[8];
  const float* ln_g   = (const float*)d_in[9];
  const float* ln_b   = (const float*)d_in[10];
  const float* proj_w = (const float*)d_in[11];
  const float* proj_b = (const float*)d_in[12];

  char* ws = (char*)d_ws;
  size_t off = 0;
  auto alloc = [&](size_t bytes) -> void* {
    void* p = ws + off;
    off += (bytes + 255) & ~(size_t)255;
    return p;
  };

  bf16* xb      = (bf16*)alloc((size_t)M_Q * C_ * 2);
  bf16* wq_p    = (bf16*)alloc((size_t)C_ * C_ * 2);        // [N][K] = torch layout
  bf16* wkv_p   = (bf16*)alloc((size_t)2 * C_ * C_ * 2);
  bf16* wproj_p = (bf16*)alloc((size_t)C_ * C_ * 2);
  bf16* wsr_p   = (bf16*)alloc((size_t)C_ * K_SR * 2);
  bf16* a_sr    = (bf16*)alloc((size_t)M_SR * K_SR * 2);
  bf16* q_bf    = (bf16*)alloc((size_t)M_Q * C_ * 2);
  float* xsr_f  = (float*)alloc((size_t)M_SR * C_ * 4);
  bf16* xsr_ln  = (bf16*)alloc((size_t)M_SR * C_ * 2);
  bf16* kv_bf   = (bf16*)alloc((size_t)M_SR * 2 * C_ * 2);
  bf16* att_o   = (bf16*)alloc((size_t)M_Q * C_ * 2);

  // weights: B operand = [N][K] row-major = torch [out][in] -> plain bf16 cast
  cvt_bf16_kernel<<<(C_ * C_) / 256, 256, 0, stream>>>(q_w, wq_p, C_ * C_);
  cvt_bf16_kernel<<<(2 * C_ * C_) / 256, 256, 0, stream>>>(kv_w, wkv_p, 2 * C_ * C_);
  cvt_bf16_kernel<<<(C_ * C_) / 256, 256, 0, stream>>>(proj_w, wproj_p, C_ * C_);
  pack_sr_kernel<<<(C_ * K_SR) / 256, 256, 0, stream>>>(sr_w, wsr_p);

  // activations
  cvt_bf16_kernel<<<(M_Q * C_) / 256, 256, 0, stream>>>(x, xb, M_Q * C_);
  im2col_kernel<<<(M_SR * K_SR) / 256, 256, 0, stream>>>(x, a_sr);

  // Q = x @ q_w^T + q_b  (bf16 out)
  gemm_bf16_kernel<1><<<dim3(M_Q / GTM, C_ / GTN), 256, 0, stream>>>(
      xb, wq_p, q_b, q_bf, M_Q, C_, C_);

  // x_sr = im2col @ sr_w + sr_b (f32 out), then LN -> bf16
  gemm_bf16_kernel<0><<<dim3(M_SR / GTM, C_ / GTN), 256, 0, stream>>>(
      a_sr, wsr_p, sr_b, xsr_f, M_SR, C_, K_SR);
  ln_kernel<<<M_SR, 256, 0, stream>>>(xsr_f, ln_g, ln_b, xsr_ln);

  // kv = x_sr @ kv_w^T + kv_b (bf16 out)
  gemm_bf16_kernel<1><<<dim3(M_SR / GTM, (2 * C_) / GTN), 256, 0, stream>>>(
      xsr_ln, wkv_p, kv_b, kv_bf, M_SR, 2 * C_, C_);

  // attention
  attn_kernel<<<dim3(N_TOK / 128, B_ * HEADS_), 256, 0, stream>>>(q_bf, kv_bf, att_o);

  // out = attn_out @ proj_w^T + proj_b (f32 to d_out)
  gemm_bf16_kernel<0><<<dim3(M_Q / GTM, C_ / GTN), 256, 0, stream>>>(
      att_o, wproj_p, proj_b, (float*)d_out, M_Q, C_, C_);

  (void)in_sizes; (void)n_in; (void)out_size; (void)ws_size;
}
